// VNResnetPointnet_57080115364769
// MI455X (gfx1250) — matrix operand order repaired and checked by reading, not compiled, not run on hardware
//
#include <hip/hip_runtime.h>

#define B_SZ  8
#define N_PTS 1024
#define M3    24576            // B * 3 * N columns
#define EPS_V 1e-6f

typedef __attribute__((ext_vector_type(16))) __bf16 v16bf;
typedef __attribute__((ext_vector_type(8)))  __bf16 v8bf;   // 16 bytes
typedef __attribute__((ext_vector_type(8)))  float  v8f;
typedef unsigned short u16;
typedef unsigned int   u32;

__device__ __forceinline__ u16 f2b(float f) {
  u32 u = __builtin_bit_cast(u32, f);
  u32 r = (u + 0x7fffu + ((u >> 16) & 1u)) >> 16;   // RNE f32 -> bf16
  return (u16)r;
}
__device__ __forceinline__ float b2f(u16 h) {
  u32 u = ((u32)h) << 16;
  return __builtin_bit_cast(float, u);
}
__device__ __forceinline__ __bf16 bbits(u16 h) {
  return __builtin_bit_cast(__bf16, h);
}
__device__ __forceinline__ float bf2f(__bf16 h) {
  return b2f(__builtin_bit_cast(u16, h));
}

// ---------------- weight f32 -> bf16 (weights stay row-major [Cout][Cin]) ----------------
__global__ void cvt_kernel(const float* __restrict__ in, u16* __restrict__ out, int n) {
  int i = blockIdx.x * 256 + threadIdx.x;
  if (i < n) out[i] = f2b(in[i]);
}

// ---------------- center point clouds ----------------
__global__ void center_kernel(const float* __restrict__ p, float* __restrict__ pc) {
  __shared__ float sx[256], sy[256], sz[256];
  int b = blockIdx.x;
  const float* pb = p + (size_t)b * N_PTS * 3;
  float ax = 0.f, ay = 0.f, az = 0.f;
  for (int n = threadIdx.x; n < N_PTS; n += 256) {
    ax += pb[n*3+0]; ay += pb[n*3+1]; az += pb[n*3+2];
  }
  sx[threadIdx.x] = ax; sy[threadIdx.x] = ay; sz[threadIdx.x] = az;
  __syncthreads();
  for (int s = 128; s > 0; s >>= 1) {
    if (threadIdx.x < s) {
      sx[threadIdx.x] += sx[threadIdx.x+s];
      sy[threadIdx.x] += sy[threadIdx.x+s];
      sz[threadIdx.x] += sz[threadIdx.x+s];
    }
    __syncthreads();
  }
  float mx = sx[0]*(1.f/N_PTS), my = sy[0]*(1.f/N_PTS), mz = sz[0]*(1.f/N_PTS);
  float* pcb = pc + (size_t)b * N_PTS * 3;
  for (int n = threadIdx.x; n < N_PTS; n += 256) {
    pcb[n*3+0] = pb[n*3+0] - mx;
    pcb[n*3+1] = pb[n*3+1] - my;
    pcb[n*3+2] = pb[n*3+2] - mz;
  }
}

// ---------------- KNN(20) -> mean neighbor position ----------------
__global__ void knn_mean_kernel(const float* __restrict__ pc, float* __restrict__ mnbr) {
  __shared__ float sp[N_PTS * 3];       // 12 KB: one batch of centered points
  int b = blockIdx.x >> 2;
  int chunk = blockIdx.x & 3;
  const float* pb = pc + (size_t)b * N_PTS * 3;
  for (int t = threadIdx.x; t < N_PTS*3; t += 256) sp[t] = pb[t];
  __syncthreads();
  int i = chunk * 256 + threadIdx.x;
  float xi = sp[i*3+0], yi = sp[i*3+1], zi = sp[i*3+2];
  float bd[20]; int bi[20];
  for (int k = 0; k < 20; ++k) { bd[k] = 3.0e38f; bi[k] = 0; }
  float curmax = 3.0e38f; int maxslot = 0;
  for (int j = 0; j < N_PTS; ++j) {
    float dx = sp[j*3+0]-xi, dy = sp[j*3+1]-yi, dz = sp[j*3+2]-zi;
    float d = dx*dx + dy*dy + dz*dz;
    if (d < curmax) {
      bd[maxslot] = d; bi[maxslot] = j;
      curmax = bd[0]; maxslot = 0;
      for (int k = 1; k < 20; ++k)
        if (bd[k] > curmax) { curmax = bd[k]; maxslot = k; }
    }
  }
  float sxm = 0.f, sym = 0.f, szm = 0.f;
  for (int k = 0; k < 20; ++k) {
    int j = bi[k];
    sxm += sp[j*3+0]; sym += sp[j*3+1]; szm += sp[j*3+2];
  }
  float inv = 1.f / 20.f;
  float* o = mnbr + (size_t)(b*N_PTS + i) * 3;
  o[0] = sxm*inv; o[1] = sym*inv; o[2] = szm*inv;
}

// ---------------- initial features (column-major: out[m*256 + c]) ----------------
__global__ void init_net_kernel(const float* __restrict__ pc, const float* __restrict__ mnbr,
                                const float* __restrict__ Wp, u16* __restrict__ out) {
  int idx = blockIdx.x * 256 + threadIdx.x;      // exact grid: M3*256 elements
  int m   = idx >> 8;                            // column (b*3+v)*N + n
  int c   = idx & 255;
  int bv  = m / N_PTS;
  int n   = m - bv * N_PTS;
  int b   = bv / 3;
  int v   = bv - b * 3;
  const float* ct = pc   + (size_t)(b*N_PTS + n) * 3;
  const float* mn = mnbr + (size_t)(b*N_PTS + n) * 3;
  int v1 = (v == 2) ? 0 : v + 1;
  int v2 = (v == 0) ? 2 : v - 1;
  float e0 = mn[v] - ct[v];                        // mean(nbr - ctr)
  float e1 = ct[v];                                // ctr
  float e2 = mn[v1]*ct[v2] - mn[v2]*ct[v1];        // cross(mean_nbr, ctr)
  float val = Wp[c*3+0]*e0 + Wp[c*3+1]*e1 + Wp[c*3+2]*e2;
  out[idx] = f2b(val);
}

// ================= WMMA GEMM kernels =================
// Column-major activations: X[m*CIN + c].  Each wave computes a 32(row)x16(col)
// tile for all three vector components: 2 A-tiles x 3 B-tiles -> 6 WMMAs per
// k-step, with B loaded once (halves L2 activation traffic vs 16-row tiles).

// ---- plain GEMM (+ optional residual add):  OUT[m*Cs + cb] = W @ X (+ ADDB) ----
template <int CIN, bool ADDC>
__global__ __launch_bounds__(128) void vn_gemm(
    const u16* __restrict__ W, const u16* __restrict__ X,
    const u16* __restrict__ ADDB, u16* __restrict__ OUT,
    int Cout, int Cs, int nTiles)
{
  int wave = blockIdx.x * 4 + (threadIdx.x >> 5);
  if (wave >= nTiles) return;                     // wave-uniform; EXEC stays all-ones
  int lane = threadIdx.x & 31;
  int rt  = Cout >> 5;            // row-pair tiles
  int crt = wave % rt;
  int bn  = wave / rt;            // (batch, n-tile): 512 total
  int b   = bn >> 6;
  int n0  = (bn & 63) << 4;
  int c0  = crt << 5;             // 32 output channels per wave
  int col = lane & 15;
  int hk  = lane >> 4;

  int mm[3];
  mm[0] = (b*3 + 0) * N_PTS + n0 + col;
  mm[1] = mm[0] + N_PTS;
  mm[2] = mm[1] + N_PTS;

  const u16* wb0 = W + (size_t)(c0 + col)      * CIN + hk * 8;
  const u16* wb1 = W + (size_t)(c0 + 16 + col) * CIN + hk * 8;

  v8f acc[2][3];
#pragma unroll
  for (int t = 0; t < 2; ++t)
#pragma unroll
    for (int v = 0; v < 3; ++v) acc[t][v] = (v8f){0,0,0,0,0,0,0,0};

#pragma unroll
  for (int k0 = 0; k0 < CIN; k0 += 32) {
    v16bf A0, A1;
    {
      v8bf lo = *(const v8bf*)(wb0 + k0);
      v8bf hi = *(const v8bf*)(wb0 + k0 + 16);
#pragma unroll
      for (int j = 0; j < 8; ++j) { A0[j] = lo[j]; A0[8+j] = hi[j]; }
    }
    {
      v8bf lo = *(const v8bf*)(wb1 + k0);
      v8bf hi = *(const v8bf*)(wb1 + k0 + 16);
#pragma unroll
      for (int j = 0; j < 8; ++j) { A1[j] = lo[j]; A1[8+j] = hi[j]; }
    }
#pragma unroll
    for (int v = 0; v < 3; ++v) {
      const u16* xp = X + (size_t)mm[v] * CIN + hk * 8 + k0;
      v8bf blo = *(const v8bf*)(xp);
      v8bf bhi = *(const v8bf*)(xp + 16);
      v16bf Bv;
#pragma unroll
      for (int j = 0; j < 8; ++j) { Bv[j] = blo[j]; Bv[8+j] = bhi[j]; }
      acc[0][v] = __builtin_amdgcn_wmma_f32_16x16x32_bf16(false, A0, false, Bv, (short)0,
                                                          acc[0][v], false, false);
      acc[1][v] = __builtin_amdgcn_wmma_f32_16x16x32_bf16(false, A1, false, Bv, (short)0,
                                                          acc[1][v], false, false);
    }
  }

  // D layout: VGPR r -> channel = cb + r (contiguous) at column mm[v]
#pragma unroll
  for (int t = 0; t < 2; ++t) {
    int cb = c0 + t*16 + 8*hk;
#pragma unroll
    for (int v = 0; v < 3; ++v) {
      float o[8];
#pragma unroll
      for (int r = 0; r < 8; ++r) o[r] = acc[t][v][r];
      if constexpr (ADDC) {
        v8bf av = *(const v8bf*)(ADDB + (size_t)mm[v] * Cout + cb);
#pragma unroll
        for (int r = 0; r < 8; ++r) o[r] += bf2f(av[r]);
      }
      v8bf ov;
#pragma unroll
      for (int r = 0; r < 8; ++r) ov[r] = bbits(f2b(o[r]));
      *(v8bf*)(OUT + (size_t)mm[v] * Cs + cb) = ov;
    }
  }
}

// ---- fused VN-LeakyReLU:  OUT = lrelu(X, W @ X)   (Cout == CIN, square W) ----
template <int CIN>
__global__ __launch_bounds__(128) void vn_gemm_lrelu(
    const u16* __restrict__ W, const u16* __restrict__ X,
    u16* __restrict__ OUT, int nTiles)
{
  int wave = blockIdx.x * 4 + (threadIdx.x >> 5);
  if (wave >= nTiles) return;
  int lane = threadIdx.x & 31;
  constexpr int rt = CIN >> 5;    // row-pair tiles (Cout == CIN)
  int crt = wave % rt;
  int bn  = wave / rt;
  int b   = bn >> 6;
  int n0  = (bn & 63) << 4;
  int c0  = crt << 5;
  int col = lane & 15;
  int hk  = lane >> 4;

  int mm[3];
  mm[0] = (b*3 + 0) * N_PTS + n0 + col;
  mm[1] = mm[0] + N_PTS;
  mm[2] = mm[1] + N_PTS;

  const u16* wb0 = W + (size_t)(c0 + col)      * CIN + hk * 8;
  const u16* wb1 = W + (size_t)(c0 + 16 + col) * CIN + hk * 8;

  v8f acc[2][3];
#pragma unroll
  for (int t = 0; t < 2; ++t)
#pragma unroll
    for (int v = 0; v < 3; ++v) acc[t][v] = (v8f){0,0,0,0,0,0,0,0};

#pragma unroll
  for (int k0 = 0; k0 < CIN; k0 += 32) {
    v16bf A0, A1;
    {
      v8bf lo = *(const v8bf*)(wb0 + k0);
      v8bf hi = *(const v8bf*)(wb0 + k0 + 16);
#pragma unroll
      for (int j = 0; j < 8; ++j) { A0[j] = lo[j]; A0[8+j] = hi[j]; }
    }
    {
      v8bf lo = *(const v8bf*)(wb1 + k0);
      v8bf hi = *(const v8bf*)(wb1 + k0 + 16);
#pragma unroll
      for (int j = 0; j < 8; ++j) { A1[j] = lo[j]; A1[8+j] = hi[j]; }
    }
#pragma unroll
    for (int v = 0; v < 3; ++v) {
      const u16* xp = X + (size_t)mm[v] * CIN + hk * 8 + k0;
      v8bf blo = *(const v8bf*)(xp);
      v8bf bhi = *(const v8bf*)(xp + 16);
      v16bf Bv;
#pragma unroll
      for (int j = 0; j < 8; ++j) { Bv[j] = blo[j]; Bv[8+j] = bhi[j]; }
      acc[0][v] = __builtin_amdgcn_wmma_f32_16x16x32_bf16(false, A0, false, Bv, (short)0,
                                                          acc[0][v], false, false);
      acc[1][v] = __builtin_amdgcn_wmma_f32_16x16x32_bf16(false, A1, false, Bv, (short)0,
                                                          acc[1][v], false, false);
    }
  }

#pragma unroll
  for (int t = 0; t < 2; ++t) {
    int cb = c0 + t*16 + 8*hk;
    v8bf xv0 = *(const v8bf*)(X + (size_t)mm[0] * CIN + cb);
    v8bf xv1 = *(const v8bf*)(X + (size_t)mm[1] * CIN + cb);
    v8bf xv2 = *(const v8bf*)(X + (size_t)mm[2] * CIN + cb);
    v8bf o0, o1, o2;
#pragma unroll
    for (int r = 0; r < 8; ++r) {
      float d0 = acc[t][0][r], d1 = acc[t][1][r], d2 = acc[t][2][r];
      float x0 = bf2f(xv0[r]), x1 = bf2f(xv1[r]), x2 = bf2f(xv2[r]);
      float dot = x0*d0 + x1*d1 + x2*d2;      // lane-local reduction over vector axis
      float dd  = d0*d0 + d1*d1 + d2*d2;
      float f = (dot < 0.f) ? (dot / (dd + EPS_V)) : 0.f;
      o0[r] = bbits(f2b(x0 - f*d0));
      o1[r] = bbits(f2b(x1 - f*d1));
      o2[r] = bbits(f2b(x2 - f*d2));
    }
    *(v8bf*)(OUT + (size_t)mm[0] * CIN + cb) = o0;
    *(v8bf*)(OUT + (size_t)mm[1] * CIN + cb) = o1;
    *(v8bf*)(OUT + (size_t)mm[2] * CIN + cb) = o2;
  }
}

// ---------------- mean over points -> broadcast into channels [128..255] (stride 256) ----------------
__global__ void pool_concat_kernel(u16* __restrict__ buf) {
  int bv = blockIdx.x;                  // 24 = B*3
  int c  = threadIdx.x;                 // 128 channels
  size_t mbase = (size_t)bv * N_PTS;
  float s = 0.f;
  for (int n = 0; n < N_PTS; ++n) s += b2f(buf[(mbase + n) * 256 + c]);
  u16 mb = f2b(s * (1.f/N_PTS));
  for (int n = 0; n < N_PTS; ++n) buf[(mbase + n) * 256 + 128 + c] = mb;
}

// ---------------- final mean over points (stride 128) -> y[b][c][v] (f32) ----------------
__global__ void pool_final_kernel(const u16* __restrict__ buf, float* __restrict__ y) {
  int bv = blockIdx.x;                  // 24
  int b = bv / 3, v = bv - b*3;
  int c = threadIdx.x;                  // 128
  size_t mbase = (size_t)bv * N_PTS;
  float s = 0.f;
  for (int n = 0; n < N_PTS; ++n) s += b2f(buf[(mbase + n) * 128 + c]);
  y[b*384 + c*3 + v] = s * (1.f/N_PTS);
}

// ---------------- head: c = W_c @ vn_lrelu(y, Wd_act) ----------------
__global__ __launch_bounds__(128) void final_kernel(
    const float* __restrict__ y, const float* __restrict__ WdA,
    const float* __restrict__ Wc, float* __restrict__ out)
{
  __shared__ float ys[384];
  __shared__ float as[384];
  int b = blockIdx.x;
  int c = threadIdx.x;                  // 128 threads
  for (int t = threadIdx.x; t < 384; t += 128) ys[t] = y[b*384 + t];
  __syncthreads();
  float d0 = 0.f, d1 = 0.f, d2 = 0.f;
  for (int ci = 0; ci < 128; ++ci) {
    float w = WdA[c*128 + ci];
    d0 += w * ys[ci*3+0]; d1 += w * ys[ci*3+1]; d2 += w * ys[ci*3+2];
  }
  float x0 = ys[c*3+0], x1 = ys[c*3+1], x2 = ys[c*3+2];
  float dot = x0*d0 + x1*d1 + x2*d2;
  float dd  = d0*d0 + d1*d1 + d2*d2;
  float f = (dot < 0.f) ? (dot / (dd + EPS_V)) : 0.f;
  as[c*3+0] = x0 - f*d0; as[c*3+1] = x1 - f*d1; as[c*3+2] = x2 - f*d2;
  __syncthreads();
  float o0 = 0.f, o1 = 0.f, o2 = 0.f;
  for (int ci = 0; ci < 128; ++ci) {
    float w = Wc[c*128 + ci];
    o0 += w * as[ci*3+0]; o1 += w * as[ci*3+1]; o2 += w * as[ci*3+2];
  }
  out[b*384 + c*3 + 0] = o0;
  out[b*384 + c*3 + 1] = o1;
  out[b*384 + c*3 + 2] = o2;
}

extern "C" void kernel_launch(void* const* d_in, const int* in_sizes, int n_in,
                              void* d_out, int out_size, void* d_ws, size_t ws_size,
                              hipStream_t stream)
{
  (void)in_sizes; (void)n_in; (void)out_size; (void)ws_size;
  const float* p     = (const float*)d_in[0];
  const float* W_pos = (const float*)d_in[1];
  const float* Wd0s  = (const float*)d_in[2];
  const float* W0s   = (const float*)d_in[3];
  const float* Wd1s  = (const float*)d_in[4];
  const float* W1s   = (const float*)d_in[5];
  const float* Wss   = (const float*)d_in[6];
  const float* Wd_a  = (const float*)d_in[7];
  const float* W_c   = (const float*)d_in[8];

  char* ws = (char*)d_ws;
  size_t off = 0;
  auto take = [&](size_t bytes) -> char* {
    off = (off + 255) & ~(size_t)255;
    char* r = ws + off;
    off += bytes;
    return r;
  };
  float* pc   = (float*)take((size_t)B_SZ*N_PTS*3*sizeof(float));
  float* mnbr = (float*)take((size_t)B_SZ*N_PTS*3*sizeof(float));
  u16* wd0b = (u16*)take((size_t)5*256*256*2);
  u16* w0b  = (u16*)take((size_t)5*128*256*2);
  u16* wd1b = (u16*)take((size_t)5*128*128*2);
  u16* w1b  = (u16*)take((size_t)5*128*128*2);
  u16* wsb  = (u16*)take((size_t)5*128*256*2);
  u16* bufA = (u16*)take((size_t)256*M3*2);
  u16* bufB = (u16*)take((size_t)256*M3*2);
  u16* a0b  = (u16*)take((size_t)256*M3*2);
  u16* t0b  = (u16*)take((size_t)128*M3*2);
  u16* a1b  = (u16*)take((size_t)128*M3*2);
  u16* dxb  = (u16*)take((size_t)128*M3*2);
  u16* lastb= (u16*)take((size_t)128*M3*2);
  float* yf = (float*)take((size_t)B_SZ*128*3*sizeof(float));

  auto cvt = [&](const float* s, u16* d, int n) {
    cvt_kernel<<<(n + 255)/256, 256, 0, stream>>>(s, d, n);
  };
  cvt(Wd0s, wd0b, 5*256*256);
  cvt(W0s,  w0b,  5*128*256);
  cvt(Wd1s, wd1b, 5*128*128);
  cvt(W1s,  w1b,  5*128*128);
  cvt(Wss,  wsb,  5*128*256);

  center_kernel<<<B_SZ, 256, 0, stream>>>(p, pc);
  knn_mean_kernel<<<B_SZ*4, 256, 0, stream>>>(pc, mnbr);
  init_net_kernel<<<(M3*256)/256, 256, 0, stream>>>(pc, mnbr, W_pos, bufA);

  u16* cur = bufA;
  u16* nxt = bufB;
  for (int i = 0; i < 5; ++i) {
    const u16* wd0 = wd0b + (size_t)i*256*256;
    const u16* w0  = w0b  + (size_t)i*128*256;
    const u16* wd1 = wd1b + (size_t)i*128*128;
    const u16* w1  = w1b  + (size_t)i*128*128;
    const u16* wsw = wsb  + (size_t)i*128*256;
    // a0 = vn_lrelu(cur, Wd0)                [256ch, stride 256]; 4096 waves
    vn_gemm_lrelu<256><<<1024, 128, 0, stream>>>(wd0, cur, a0b, 4096);
    // t0 = W0 @ a0                           [128ch, stride 128]; 2048 waves
    vn_gemm<256, false><<<512, 128, 0, stream>>>(w0, a0b, nullptr, t0b, 128, 128, 2048);
    // a1 = vn_lrelu(t0, Wd1)                 [128ch, stride 128]; 2048 waves
    vn_gemm_lrelu<128><<<512, 128, 0, stream>>>(wd1, t0b, a1b, 2048);
    // dx = W1 @ a1                           [128ch, stride 128]; 2048 waves
    vn_gemm<128, false><<<512, 128, 0, stream>>>(w1, a1b, nullptr, dxb, 128, 128, 2048);
    if (i < 4) {
      // out = Ws @ cur + dx -> channels [0..127] of nxt (stride 256)
      vn_gemm<256, true><<<512, 128, 0, stream>>>(wsw, cur, dxb, nxt, 128, 256, 2048);
      pool_concat_kernel<<<24, 128, 0, stream>>>(nxt);
      u16* t = cur; cur = nxt; nxt = t;
    } else {
      // final block output -> stride-128 buffer, then pool over points
      vn_gemm<256, true><<<512, 128, 0, stream>>>(wsw, cur, dxb, lastb, 128, 128, 2048);
      pool_final_kernel<<<24, 128, 0, stream>>>(lastb, yf);
    }
  }
  final_kernel<<<B_SZ, 128, 0, stream>>>(yf, Wd_a, W_c, (float*)d_out);
}
